// ManualLSTM_10806137717345
// MI455X (gfx1250) — matrix operand order repaired and checked by reading, not compiled, not run on hardware
//
#include <hip/hip_runtime.h>
#include <stdint.h>

// LSTM: B=64, T=512, I=512, H=1024, gates=4H=4096
#define B_   64
#define T_   512
#define I_   512
#define H_   1024
#define G4   4096
#define KTOT 1536              // I + H
#define NKCH 48                // KTOT/32 K-chunks
#define WSW_CHUNK (256*32*16)  // uint16 elems per kchunk slab (256 coltiles)
#define WSW_GATE  (64*32*16)   // uint16 elems per gate group within a slab

typedef __attribute__((ext_vector_type(16))) __bf16 v16bf;
typedef __attribute__((ext_vector_type(8)))  float  v8f;

__device__ __forceinline__ uint16_t f2bf(float f) {
    // round-to-nearest-even f32 -> bf16
    uint32_t u = __float_as_uint(f);
    uint32_t r = u + 0x7FFFu + ((u >> 16) & 1u);
    return (uint16_t)(r >> 16);
}
__device__ __forceinline__ uint32_t pack2bf(float a, float b) {
    return (uint32_t)f2bf(a) | ((uint32_t)f2bf(b) << 16);
}
__device__ __forceinline__ v16bf make_frag(uint4 lo, uint4 hi) {
    union { uint4 q[2]; v16bf v; } u;
    u.q[0] = lo; u.q[1] = hi;
    return u.v;
}
__device__ __forceinline__ float fsig(float x) {
    return __builtin_amdgcn_rcpf(1.0f + __expf(-x));
}
__device__ __forceinline__ float ftanh(float x) {
    // saturates correctly at +/-1 for large |x|
    return 1.0f - 2.0f * __builtin_amdgcn_rcpf(__expf(2.0f * x) + 1.0f);
}

// ---------------------------------------------------------------------------
// Prep 1: swizzle [W_in; W_h] (f32) into bf16 WMMA B-fragment order:
//   Wsw[kchunk][coltile][lane][s], s=0..15:  element (k, col) with
//   kchunk=k>>5, kk=k&31, coltile=col>>4, n=col&15, lane=(kk&16)+n, s=kk&15.
// One thread produces one lane-block (16 bf16 = 32 contiguous bytes).
// ---------------------------------------------------------------------------
__global__ __launch_bounds__(256)
void lstm_prep_w(const float* __restrict__ Win, const float* __restrict__ Wh,
                 uint16_t* __restrict__ Wsw) {
    int tid16 = blockIdx.x * blockDim.x + threadIdx.x;   // 0 .. 393215
    if (tid16 >= NKCH * 256 * 32) return;
    int lane = tid16 & 31;
    int ct   = (tid16 >> 5) & 255;
    int kch  = tid16 >> 13;
    int col  = ct * 16 + (lane & 15);
    int kb   = kch * 32 + (lane & 16);
    union { uint16_t h[16]; uint4 q[2]; } u;
    #pragma unroll
    for (int j = 0; j < 16; ++j) {
        int k = kb + j;
        float v = (k < I_) ? Win[k * G4 + col] : Wh[(k - I_) * G4 + col];
        u.h[j] = f2bf(v);
    }
    uint4* dst = (uint4*)(Wsw + (size_t)tid16 * 16);
    dst[0] = u.q[0];
    dst[1] = u.q[1];
}

// ---------------------------------------------------------------------------
// Prep 2: convert + transpose x (B,T,I) f32 -> xT (T,B,I) bf16 (8 elems/thr),
// zero h0 (bf16) and c0 (f32) state.
// ---------------------------------------------------------------------------
__global__ __launch_bounds__(256)
void lstm_prep_x(const float* __restrict__ x, uint16_t* __restrict__ xT,
                 uint16_t* __restrict__ h0, float* __restrict__ c0) {
    const int NX8 = B_ * T_ * I_ / 8;          // 2,097,152
    const int NH8 = B_ * H_ / 8;               // 8,192 (h0, bf16 x8)
    const int NC4 = B_ * H_ / 4;               // 16,384 (c0, f32 x4)
    int idx = blockIdx.x * blockDim.x + threadIdx.x;
    if (idx < NX8) {
        int i8 = idx & 63;                     // i / 8
        int t  = (idx >> 6) & 511;
        int b  = idx >> 15;
        const float4* src = (const float4*)(x + ((size_t)(b * T_ + t) * I_ + i8 * 8));
        float4 a = src[0], bb = src[1];
        uint4 o;
        o.x = pack2bf(a.x, a.y);   o.y = pack2bf(a.z, a.w);
        o.z = pack2bf(bb.x, bb.y); o.w = pack2bf(bb.z, bb.w);
        *(uint4*)(xT + ((size_t)(t * B_ + b) * I_ + i8 * 8)) = o;
    } else if (idx < NX8 + NH8) {
        uint4 z = {0, 0, 0, 0};
        *(uint4*)(h0 + (size_t)(idx - NX8) * 8) = z;
    } else if (idx < NX8 + NH8 + NC4) {
        float4 z = {0.f, 0.f, 0.f, 0.f};
        *(float4*)(c0 + (size_t)(idx - NX8 - NH8) * 4) = z;
    }
}

// ---------------------------------------------------------------------------
// One LSTM time step, fully fused, zero LDS / zero barriers:
//   gates = [x_t | h_in] @ W + bias  (bf16 WMMA, f32 acc), fragments loaded
//   straight from global (weights pre-swizzled, A pre-converted to bf16),
//   software-pipelined: chunk k+1's fragments are loaded while chunk k's
//   four WMMAs execute.
// Grid: 64 blocks (16 h-cols each).  Block: 128 = 4 wave32s.
// Wave w owns M-tile w (16 batch rows) and ALL 4 gate tiles -> the cell
// update (sigmoid/tanh/c/h) runs entirely in registers per lane.
// ---------------------------------------------------------------------------
__global__ __launch_bounds__(128)
void lstm_step(const uint16_t* __restrict__ xT, const float* __restrict__ bias,
               const uint16_t* __restrict__ Wsw,
               const uint16_t* __restrict__ h_in, uint16_t* __restrict__ h_out,
               float* __restrict__ c_st, float* __restrict__ out, int t)
{
    const int tid    = threadIdx.x;
    const int lane   = tid & 31;
    const int wid    = tid >> 5;          // M-tile 0..3
    const int hc0    = blockIdx.x * 16;
    const int nloc   = lane & 15;
    const int hiHalf = lane >> 4;

    // A-fragment addressing (this code's k order, matching Wsw: lanes 0-15 K 0..15,
    // lanes 16-31 K 16..31; two contiguous 16B pieces per lane)
    const int row = wid * 16 + nloc;
    const int akh = hiHalf * 16;
    const uint16_t* xrow = xT   + (size_t)(t * B_ + row) * I_ + akh;
    const uint16_t* hrow = h_in + (size_t)row * H_ + akh;

    // B-fragment base for this block/lane; + kch*WSW_CHUNK + g*WSW_GATE
    const uint16_t* bbase = Wsw + ((size_t)(blockIdx.x * 32 + lane)) * 16;

    auto loadA = [&](int kch) -> v16bf {
        const uint16_t* ab = (kch < 16) ? (xrow + kch * 32)
                                        : (hrow + (kch - 16) * 32);
        return make_frag(*(const uint4*)ab, *(const uint4*)(ab + 8));
    };
    auto loadB = [&](int kch, int g) -> v16bf {
        const uint16_t* bp = bbase + (size_t)kch * WSW_CHUNK
                                   + (size_t)g   * WSW_GATE;
        return make_frag(*(const uint4*)bp, *(const uint4*)(bp + 8));
    };

    v8f acc[4] = {{}, {}, {}, {}};

    // software pipeline: prefetch chunk k+1 while chunk k's WMMAs run
    v16bf aCur = loadA(0);
    v16bf bCur0 = loadB(0, 0), bCur1 = loadB(0, 1),
          bCur2 = loadB(0, 2), bCur3 = loadB(0, 3);

    #pragma unroll 2
    for (int kch = 0; kch < NKCH; ++kch) {
        const int nk = (kch + 1 < NKCH) ? kch + 1 : NKCH - 1;  // clamp (1 dup load)
        v16bf aNxt  = loadA(nk);
        v16bf bNxt0 = loadB(nk, 0), bNxt1 = loadB(nk, 1),
              bNxt2 = loadB(nk, 2), bNxt3 = loadB(nk, 3);

        acc[0] = __builtin_amdgcn_wmma_f32_16x16x32_bf16(
                     false, aCur, false, bCur0, (short)0, acc[0], false, false);
        acc[1] = __builtin_amdgcn_wmma_f32_16x16x32_bf16(
                     false, aCur, false, bCur1, (short)0, acc[1], false, false);
        acc[2] = __builtin_amdgcn_wmma_f32_16x16x32_bf16(
                     false, aCur, false, bCur2, (short)0, acc[2], false, false);
        acc[3] = __builtin_amdgcn_wmma_f32_16x16x32_bf16(
                     false, aCur, false, bCur3, (short)0, acc[3], false, false);

        aCur = aNxt;
        bCur0 = bNxt0; bCur1 = bNxt1; bCur2 = bNxt2; bCur3 = bNxt3;
    }

    // Fused cell update, all in registers: C layout => lane holds N=nloc,
    // VGPR r holds M = hiHalf*8 + r, identical slot across the 4 gate accs.
    const int hc = hc0 + nloc;
    const float bi = bias[0 * H_ + hc];
    const float bf = bias[1 * H_ + hc];
    const float bg = bias[2 * H_ + hc];
    const float bo = bias[3 * H_ + hc];
    const int HSOFF = B_ * T_ * H_;

    #pragma unroll
    for (int r = 0; r < 8; ++r) {
        int b = wid * 16 + hiHalf * 8 + r;
        float iv = fsig (acc[0][r] + bi);
        float fv = fsig (acc[1][r] + bf);
        float gv = ftanh(acc[2][r] + bg);
        float ov = fsig (acc[3][r] + bo);
        float c_old = c_st[b * H_ + hc];
        float c_new = fv * c_old + iv * gv;
        float h_new = ov * ftanh(c_new);
        c_st [b * H_ + hc] = c_new;
        h_out[b * H_ + hc] = f2bf(h_new);
        out[(size_t)(b * T_ + t) * H_ + hc] = h_new;
        if (t == T_ - 1) {
            out[HSOFF + b * H_ + hc]           = h_new;  // h_T
            out[HSOFF + B_ * H_ + b * H_ + hc] = c_new;  // c_T
        }
    }
}

// ---------------------------------------------------------------------------
extern "C" void kernel_launch(void* const* d_in, const int* in_sizes, int n_in,
                              void* d_out, int out_size, void* d_ws, size_t ws_size,
                              hipStream_t stream) {
    (void)in_sizes; (void)n_in; (void)out_size; (void)ws_size;
    const float* x    = (const float*)d_in[0];   // (B,T,I)
    const float* Win  = (const float*)d_in[1];   // (I,4H)
    const float* Wh   = (const float*)d_in[2];   // (H,4H)
    const float* bias = (const float*)d_in[3];   // (4H,)
    float* out = (float*)d_out;

    // workspace layout (bytes):
    //   Wsw bf16 (swizzled weights) : 1536*4096*2 = 12,582,912
    //   xT  bf16 (T,B,I)            : 512*64*512*2 = 33,554,432
    //   h ping (bf16)               : 64*1024*2 = 131,072
    //   h pong (bf16)               : 64*1024*2 = 131,072
    //   c state (f32)               : 64*1024*4 = 262,144
    char* ws = (char*)d_ws;
    uint16_t* Wsw = (uint16_t*)ws;
    uint16_t* xT  = (uint16_t*)(ws + 12582912);
    uint16_t* h0  = (uint16_t*)(ws + 12582912 + 33554432);
    uint16_t* h1  = (uint16_t*)(ws + 12582912 + 33554432 + 131072);
    float*    c   = (float*)   (ws + 12582912 + 33554432 + 262144);

    {
        int total  = NKCH * 256 * 32;                 // 393,216 lane-blocks
        lstm_prep_w<<<(total + 255) / 256, 256, 0, stream>>>(Win, Wh, Wsw);
    }
    {
        int total = B_ * T_ * I_ / 8 + B_ * H_ / 8 + B_ * H_ / 4;
        lstm_prep_x<<<(total + 255) / 256, 256, 0, stream>>>(x, xT, h0, c);
    }

    uint16_t* hb[2] = { h0, h1 };
    for (int t = 0; t < T_; ++t) {
        lstm_step<<<64, 128, 0, stream>>>(xT, bias, Wsw,
                                          hb[t & 1], hb[(t + 1) & 1],
                                          c, out, t);
    }
}
